// TernaryLinear_5368709120113
// MI455X (gfx1250) — matrix operand order repaired and checked by reading, not compiled, not run on hardware
//
#include <hip/hip_runtime.h>
#include <stdint.h>

typedef __attribute__((ext_vector_type(16))) __bf16 v16bf;
typedef __attribute__((ext_vector_type(8)))  float  v8f;
typedef __attribute__((ext_vector_type(2)))  float  f32x2;
typedef __attribute__((ext_vector_type(2)))  __bf16 bf16x2;

#define THRESH 0.05f
#define BM 128
#define BN 128
#define BK 32
#define LDT 40          // LDS row pitch in bf16 elements (80 bytes -> bank spread)
#define KCHUNKS 16

#if defined(__gfx1250__) && __has_builtin(__builtin_amdgcn_tensor_load_to_lds) && __has_builtin(__builtin_amdgcn_s_wait_tensorcnt)
#define USE_TDM 1
#else
#define USE_TDM 0
#endif

// ---------------- prepass: ternarize w (K x N, f32) -> wtT (N x K, bf16 {-1,0,+1}) ----------------
__global__ void ternarize_kernel(const float* __restrict__ w,
                                 unsigned short* __restrict__ wtT,
                                 float* __restrict__ partial,
                                 int K, int O, int kchunk) {
  int col = blockIdx.x * blockDim.x + threadIdx.x;
  if (col >= O) return;
  int k0 = blockIdx.y * kchunk;
  float s = 0.0f;
  for (int k = k0; k < k0 + kchunk; ++k) {
    float v  = w[(size_t)k * O + col];
    float av = fabsf(v);
    s += av;
    unsigned short t = (av < THRESH) ? (unsigned short)0
                     : (v > 0.0f ? (unsigned short)0x3F80 : (unsigned short)0xBF80);
    wtT[(size_t)col * K + k] = t;   // bf16 -1/0/+1 bit patterns
  }
  partial[(size_t)blockIdx.y * O + col] = s;
}

__global__ void reduce_scale_kernel(const float* __restrict__ partial,
                                    float* __restrict__ scaleSum, int O) {
  int col = blockIdx.x * blockDim.x + threadIdx.x;
  if (col >= O) return;
  float s = 0.0f;
  for (int c = 0; c < KCHUNKS; ++c) s += partial[(size_t)c * O + col]; // fixed order: deterministic
  scaleSum[col] = s;
}

// ---------------- main bf16 WMMA GEMM ----------------
// pair f32 -> packed bf16: vector fptrunc lets the backend pick v_cvt_pk_bf16_f32
__device__ __forceinline__ uint32_t pack_bf16_2(float a, float b) {
  f32x2 f = {a, b};
  bf16x2 h = __builtin_convertvector(f, bf16x2);
  return __builtin_bit_cast(uint32_t, h);
}

struct alignas(16) Smem {
  unsigned short A[2][BM * LDT];   // x tile, bf16, double buffered
  unsigned short B[2][BN * LDT];   // ternary-w tile, bf16, double buffered
};

typedef unsigned int u32x4v __attribute__((ext_vector_type(4)));
typedef int          i32x8v __attribute__((ext_vector_type(8)));
typedef int          i32x4v __attribute__((ext_vector_type(4)));

__global__ __launch_bounds__(256) void ternary_gemm_kernel(
    const float* __restrict__ x, const unsigned short* __restrict__ wtT,
    const float* __restrict__ scaleSum, const float* __restrict__ bias,
    float* __restrict__ out, int M, int N, int K)
{
  __shared__ Smem sm;
  const int tid    = threadIdx.x;
  const int lane   = tid & 31;
  const int wid    = tid >> 5;
  const int wm     = wid & 3;      // 4 waves along M (32 rows each)
  const int wn     = wid >> 2;     // 2 waves along N (64 cols each)
  const int laneLo = lane & 15;
  const int laneHi = lane >> 4;
  const int bm     = blockIdx.y * BM;
  const int bn     = blockIdx.x * BN;
  const int NT     = K / BK;

  const int srow  = tid >> 1;      // 0..127 staging row
  const int shalf = tid & 1;       // 16-element half of the 32-wide K tile

  // ---- stage x tile: global f32 -> bf16 -> LDS ----
  auto stageA = [&](int buf, int kt) {
    const float* src = x + (size_t)(bm + srow) * K + (size_t)kt * BK + shalf * 16;
    float4 f0 = *(const float4*)(src + 0);
    float4 f1 = *(const float4*)(src + 4);
    float4 f2 = *(const float4*)(src + 8);
    float4 f3 = *(const float4*)(src + 12);
    uint4 p0, p1;
    p0.x = pack_bf16_2(f0.x, f0.y); p0.y = pack_bf16_2(f0.z, f0.w);
    p0.z = pack_bf16_2(f1.x, f1.y); p0.w = pack_bf16_2(f1.z, f1.w);
    p1.x = pack_bf16_2(f2.x, f2.y); p1.y = pack_bf16_2(f2.z, f2.w);
    p1.z = pack_bf16_2(f3.x, f3.y); p1.w = pack_bf16_2(f3.z, f3.w);
    uint4* dst = (uint4*)&sm.A[buf][srow * LDT + shalf * 16];
    dst[0] = p0;
    dst[1] = p1;
  };

#if USE_TDM
  // ---- B tile via Tensor Data Mover: one DMA per workgroup, issued by wave 0 ----
  auto issueB = [&](int buf, int kt) {
    if (wid != 0) return;
    uint64_t ga = (uint64_t)(uintptr_t)(wtT + (size_t)bn * K + (size_t)kt * BK);
    uint32_t ldsOff = (uint32_t)__builtin_offsetof(Smem, B) + (uint32_t)buf * (BN * LDT * 2);
    u32x4v g0;
    g0.x = 1u;                                           // count=1 valid descriptor
    g0.y = ldsOff;                                       // lds_addr (bytes)
    g0.z = (uint32_t)ga;                                 // global_addr[31:0]
    g0.w = (uint32_t)((ga >> 32) & 0x01FFFFFFu) | (2u << 30);  // global_addr[56:32] | type=2
    i32x8v g1;
    g1[0] = (int)((1u << 16)       // data_size = 2 bytes
                | (1u << 20)       // pad_enable
                | (3u << 22)       // pad_interval: 16 DWORDs = 64B (one 32-elem bf16 row)
                | (3u << 25));     // pad_amount: 4 DWORDs = 16B  -> 80B LDS pitch
    g1[1] = (int)(((uint32_t)K & 0xFFFFu) << 16);                                    // tensor_dim0 lo16
    g1[2] = (int)((((uint32_t)K >> 16) & 0xFFFFu) | (((uint32_t)N & 0xFFFFu) << 16));// dim0 hi | dim1 lo
    g1[3] = (int)((((uint32_t)N >> 16) & 0xFFFFu) | ((uint32_t)BK << 16));           // dim1 hi | tile_dim0=32
    g1[4] = (int)BN;               // tile_dim1 = 128 rows
    g1[5] = (int)K;                // tensor_dim0_stride lo32
    g1[6] = 0;
    g1[7] = 0;
    i32x4v gz = {0, 0, 0, 0};
#if __clang_major__ >= 23
    i32x8v gz8 = {0, 0, 0, 0, 0, 0, 0, 0};
    __builtin_amdgcn_tensor_load_to_lds(g0, g1, gz, gz, gz8, 0);
#else
    __builtin_amdgcn_tensor_load_to_lds(g0, g1, gz, gz, 0);
#endif
  };
#else
  auto issueB = [&](int buf, int kt) {
    const uint4* src = (const uint4*)(wtT + (size_t)(bn + srow) * K + (size_t)kt * BK + shalf * 16);
    uint4 q0 = src[0];
    uint4 q1 = src[1];
    uint4* dst = (uint4*)&sm.B[buf][srow * LDT + shalf * 16];
    dst[0] = q0;
    dst[1] = q1;
  };
#endif

  v8f acc[2][4] = {};

  union AB { v16bf v; uint4 q[2]; };

  auto compute = [&](int buf) {
    AB bf[4];
#pragma unroll
    for (int ni = 0; ni < 4; ++ni) {
      // B frag: lane n (0..15) = col, 16 contiguous K per lane-half
      const unsigned short* p = &sm.B[buf][(wn * 64 + ni * 16 + laneLo) * LDT + laneHi * 16];
      bf[ni].q[0] = *(const uint4*)(p);
      bf[ni].q[1] = *(const uint4*)(p + 8);
    }
#pragma unroll
    for (int mi = 0; mi < 2; ++mi) {
      // A frag: lane row, 16 contiguous K per lane-half
      AB af;
      const unsigned short* p = &sm.A[buf][(wm * 32 + mi * 16 + laneLo) * LDT + laneHi * 16];
      af.q[0] = *(const uint4*)(p);
      af.q[1] = *(const uint4*)(p + 8);
#pragma unroll
      for (int ni = 0; ni < 4; ++ni) {
        acc[mi][ni] = __builtin_amdgcn_wmma_f32_16x16x32_bf16(
            false, af.v, false, bf[ni].v, (short)0, acc[mi][ni], false, false);
      }
    }
  };

  // ---- double-buffered main loop: TDM issued first for max latency cover ----
  issueB(0, 0);
  stageA(0, 0);
  for (int kt = 0; kt < NT; ++kt) {
    int buf = kt & 1;
#if USE_TDM
    if (wid == 0) __builtin_amdgcn_s_wait_tensorcnt(0);
#endif
    __syncthreads();                       // tile kt visible to all waves
    if (kt + 1 < NT) {
      issueB(buf ^ 1, kt + 1);
      stageA(buf ^ 1, kt + 1);
    }
    compute(buf);
  }

  // ---- epilogue: y = scale[n]*acc + b[n]; streaming NT stores keep L2 for x/w ----
#pragma unroll
  for (int ni = 0; ni < 4; ++ni) {
    int c = bn + wn * 64 + ni * 16 + laneLo;
    float s  = scaleSum[c] * (1.0f / (float)K);
    float bb = bias[c];
#pragma unroll
    for (int mi = 0; mi < 2; ++mi) {
      int r0 = bm + wm * 32 + mi * 16 + laneHi * 8;
#pragma unroll
      for (int j = 0; j < 8; ++j) {
        __builtin_nontemporal_store(acc[mi][ni][j] * s + bb,
                                    out + (size_t)(r0 + j) * N + c);
      }
    }
  }
}

// ---------------- launcher ----------------
extern "C" void kernel_launch(void* const* d_in, const int* in_sizes, int n_in,
                              void* d_out, int out_size, void* d_ws, size_t ws_size,
                              hipStream_t stream) {
  const float* x = (const float*)d_in[0];
  const float* w = (const float*)d_in[1];
  const float* b = (const float*)d_in[2];
  float* out = (float*)d_out;

  const int O = in_sizes[2];          // 4096 output cols
  const int D = in_sizes[1] / O;      // 4096 reduction dim
  const int M = in_sizes[0] / D;      // 16384 rows

  unsigned short* wtT = (unsigned short*)d_ws;                         // N x K bf16 (32 MB)
  char* p = (char*)d_ws + (size_t)D * O * sizeof(unsigned short);
  float* partial  = (float*)p;                                         // KCHUNKS x O partial sums
  float* scaleSum = (float*)(p + (size_t)KCHUNKS * O * sizeof(float)); // O sums of |w|

  ternarize_kernel<<<dim3((O + 255) / 256, KCHUNKS), 256, 0, stream>>>(
      w, wtT, partial, D, O, D / KCHUNKS);
  reduce_scale_kernel<<<(O + 255) / 256, 256, 0, stream>>>(partial, scaleSum, O);
  ternary_gemm_kernel<<<dim3(O / BN, M / BM), 256, 0, stream>>>(
      x, wtT, scaleSum, b, out, M, O, D);
}
